// MinLSTM_22273700397171
// MI455X (gfx1250) — compile-verified, matrix operand order checked by reading, not run on hardware
//
#include <hip/hip_runtime.h>
#include <hip/hip_bf16.h>

// ---------------- problem constants ----------------
constexpr int B = 8, T = 2048, D = 1024, H = 1024;
constexpr int M = B * T;            // 16384 rows of x
constexpr int BM = 128, BN = 64, BK = 32;
constexpr int LDSROW = 40;          // padded row stride (halfwords): 80B = 5*16B, conflict-free
constexpr int NIT = D / BK;         // 32 K-iterations
constexpr int NC = 16;              // scan chunks
constexpr int CL = T / NC;          // chunk length = 128

constexpr int XTILE_HW = BM * LDSROW;                 // 5120 halfwords
constexpr int WTILE_HW = BN * LDSROW;                 // 2560 halfwords
constexpr int BUF_HW   = XTILE_HW + 3 * WTILE_HW;     // 12800 halfwords = 25600 B

typedef __attribute__((ext_vector_type(16))) __bf16 bf16x16;
typedef __attribute__((ext_vector_type(8)))  float  floatx8;

union Frag { uint4 u[2]; bf16x16 v; };

// fp32 -> bf16 bits, round-to-nearest-even
static __device__ __forceinline__ unsigned short f2bf(float f) {
    unsigned int u = __float_as_uint(f);
    u = (u + 0x7FFFu + ((u >> 16) & 1u)) >> 16;
    return (unsigned short)u;
}

// fast branchless softplus / log-gate helpers (hardware transcendentals)
static __device__ __forceinline__ float sp(float x) {
    return fmaxf(x, 0.f) + __logf(1.f + __expf(-fabsf(x)));
}
static __device__ __forceinline__ float logg(float z) {
    return (z >= 0.f) ? __logf(z + 0.5f) : -sp(-z);
}
static __device__ __forceinline__ float logaddexp(float x, float y) {
    return fmaxf(x, y) + __logf(1.f + __expf(-fabsf(x - y)));
}

// ---------------- async global->LDS copy (CDNA5), with sync fallback -------
#if defined(__has_builtin)
#if __has_builtin(__builtin_amdgcn_global_load_async_to_lds_b128)
#define HAVE_ASYNC_LDS 1
#endif
#endif

#ifdef HAVE_ASYNC_LDS
typedef int v4i_vs __attribute__((vector_size(4 * sizeof(int))));
typedef __attribute__((address_space(1))) v4i_vs* g_v4i_ptr;
typedef __attribute__((address_space(3))) v4i_vs* l_v4i_ptr;
#endif

static __device__ __forceinline__ void copy16_g2l(const unsigned short* g,
                                                  unsigned short* l) {
#ifdef HAVE_ASYNC_LDS
    __builtin_amdgcn_global_load_async_to_lds_b128(
        (g_v4i_ptr)(uintptr_t)g,
        (l_v4i_ptr)(unsigned)(uintptr_t)l, 0, 0);
#else
    *(uint4*)l = *(const uint4*)g;
#endif
}
static __device__ __forceinline__ void wait_async_le5() {
#ifdef HAVE_ASYNC_LDS
    asm volatile("s_wait_asynccnt 0x5" ::: "memory");
#endif
}
static __device__ __forceinline__ void wait_async_le0() {
#ifdef HAVE_ASYNC_LDS
    asm volatile("s_wait_asynccnt 0x0" ::: "memory");
#endif
}

// Load one 16x32 bf16 WMMA fragment (A or mirror B layout) from an LDS tile.
static __device__ __forceinline__ bf16x16 load_frag(const unsigned short* lds,
                                                    int row0, int lane) {
    const int r    = row0 + (lane & 15);
    const int koff = (lane < 16) ? 0 : 8;
    Frag f;
    f.u[0] = *(const uint4*)(lds + r * LDSROW + koff);
    f.u[1] = *(const uint4*)(lds + r * LDSROW + koff + 16);
    return f.v;
}

#define WMMA_BF16(A, Bm, C) \
    __builtin_amdgcn_wmma_f32_16x16x32_bf16(false, (A), false, (Bm), (short)0, (C), false, false)

// ---------------------------------------------------------------------------
// Kernel 0: fp32 -> bf16 pre-conversion (one shot; removes conversion from
// the GEMM hot loop and halves x read traffic).
// ---------------------------------------------------------------------------
__global__ __launch_bounds__(256)
void convert_bf16(const float* __restrict__ src, unsigned short* __restrict__ dst) {
    const size_t i = (size_t)(blockIdx.x * 256 + threadIdx.x) * 8;
    const float4 a = *(const float4*)(src + i);
    const float4 b = *(const float4*)(src + i + 4);
    uint4 o;
    o.x = (unsigned)f2bf(a.x) | ((unsigned)f2bf(a.y) << 16);
    o.y = (unsigned)f2bf(a.z) | ((unsigned)f2bf(a.w) << 16);
    o.z = (unsigned)f2bf(b.x) | ((unsigned)f2bf(b.y) << 16);
    o.w = (unsigned)f2bf(b.z) | ((unsigned)f2bf(b.w) << 16);
    *(uint4*)(dst + i) = o;
}

// Issue all async copies for one k-slab into one LDS buffer (5 instrs/wave).
static __device__ __forceinline__ void issue_tiles(
        const unsigned short* __restrict__ xb,
        const unsigned short* __restrict__ wfb,
        const unsigned short* __restrict__ wib,
        const unsigned short* __restrict__ whb,
        unsigned short* buf, int m0, int n0, int k0, int t) {
    unsigned short* ldsX  = buf;
    unsigned short* ldsWf = buf + XTILE_HW;
    unsigned short* ldsWi = buf + XTILE_HW + WTILE_HW;
    unsigned short* ldsWh = buf + XTILE_HW + 2 * WTILE_HW;
    // x tile: 128x32 bf16 = 512 16B chunks -> 2 per thread
#pragma unroll
    for (int p = 0; p < 2; ++p) {
        const int id = t + p * 256;
        const int rr = id >> 2, c8 = (id & 3) * 8;
        copy16_g2l(xb + (size_t)(m0 + rr) * D + k0 + c8, ldsX + rr * LDSROW + c8);
    }
    // W tiles: 64x32 bf16 = 256 chunks each -> 1 per thread per matrix
    const int rr = t >> 2, c8 = (t & 3) * 8;
    const size_t go = (size_t)(n0 + rr) * D + k0 + c8;
    const int lo = rr * LDSROW + c8;
    copy16_g2l(wfb + go, ldsWf + lo);
    copy16_g2l(wib + go, ldsWi + lo);
    copy16_g2l(whb + go, ldsWh + lo);
}

// ---------------------------------------------------------------------------
// Kernel 1: three fused bf16 GEMMs + gate-math epilogue.
// 128x64 block tile, double-buffered async global->LDS pipeline.
// ---------------------------------------------------------------------------
__global__ __launch_bounds__(256)
void minlstm_gemm_gate(const unsigned short* __restrict__ xb,
                       const unsigned short* __restrict__ wfb,
                       const unsigned short* __restrict__ wib,
                       const unsigned short* __restrict__ whb,
                       const float* __restrict__ bf_,
                       const float* __restrict__ bi_,
                       const float* __restrict__ bh_,
                       float* __restrict__ logF,
                       float* __restrict__ logV) {
    __shared__ unsigned short lds[2][BUF_HW];   // 2 x 25600 B

    const int t    = threadIdx.x;
    const int lane = t & 31;
    const int wave = t >> 5;          // 0..7
    const int mw   = wave & 3;        // M-subtiles {mw, mw+4}
    const int nw   = wave >> 2;       // N-subtiles {2nw, 2nw+1}

    const int m0 = blockIdx.x * BM;
    const int n0 = blockIdx.y * BN;

    floatx8 accF[2][2] = {}, accI[2][2] = {}, accH[2][2] = {};  // [mi][s]

    issue_tiles(xb, wfb, wib, whb, lds[0], m0, n0, 0, t);

    for (int i = 0; i < NIT; ++i) {
        const unsigned short* cur = lds[i & 1];
        if (i + 1 < NIT) {
            issue_tiles(xb, wfb, wib, whb, lds[(i + 1) & 1], m0, n0, (i + 1) * BK, t);
            wait_async_le5();         // retire current tile's group (in-order)
        } else {
            wait_async_le0();
        }
        __syncthreads();              // current tile visible to all waves

        const unsigned short* cX  = cur;
        const unsigned short* cWf = cur + XTILE_HW;
        const unsigned short* cWi = cur + XTILE_HW + WTILE_HW;
        const unsigned short* cWh = cur + XTILE_HW + 2 * WTILE_HW;

        const bf16x16 a0 = load_frag(cX, mw * 16, lane);
        const bf16x16 a1 = load_frag(cX, (mw + 4) * 16, lane);
#pragma unroll
        for (int s = 0; s < 2; ++s) {
            const int nr = (nw * 2 + s) * 16;
            const bf16x16 bF = load_frag(cWf, nr, lane);
            const bf16x16 bI = load_frag(cWi, nr, lane);
            const bf16x16 bH = load_frag(cWh, nr, lane);
            accF[0][s] = WMMA_BF16(a0, bF, accF[0][s]);
            accI[0][s] = WMMA_BF16(a0, bI, accI[0][s]);
            accH[0][s] = WMMA_BF16(a0, bH, accH[0][s]);
            accF[1][s] = WMMA_BF16(a1, bF, accF[1][s]);
            accI[1][s] = WMMA_BF16(a1, bI, accI[1][s]);
            accH[1][s] = WMMA_BF16(a1, bH, accH[1][s]);
        }
        __syncthreads();              // done reading before buffer is rewritten
    }

    // Epilogue: C/D layout -> lane holds N = lane&15, VGPR r holds M = r + 8*(lane>=16)
    const int lane15 = lane & 15;
    const int mofs   = (lane >> 4) * 8;
#pragma unroll
    for (int s = 0; s < 2; ++s) {
        const int n  = n0 + (nw * 2 + s) * 16 + lane15;
        const float bFv = bf_[n], bIv = bi_[n], bHv = bh_[n];
#pragma unroll
        for (int mi = 0; mi < 2; ++mi) {
            const int mbase = m0 + (mw + 4 * mi) * 16 + mofs;
#pragma unroll
            for (int r = 0; r < 8; ++r) {
                const int m = mbase + r;
                const float fp = accF[mi][s][r] + bFv;
                const float ip = accI[mi][s][r] + bIv;
                const float hp = accH[mi][s][r] + bHv;
                const float diff = sp(-fp) - sp(-ip);
                const float lf = -sp(diff);               // log f-gate
                const float lv = lf + diff + logg(hp);    // log_i + log g(h_pre)
                const size_t o = (size_t)m * H + n;
                logF[o] = lf;
                logV[o] = lv;
            }
        }
    }
}

// ---------------------------------------------------------------------------
// Scan pass A: per (b,h,chunk) compute chunkA = sum(log_f), chunkL = local LSE.
// ---------------------------------------------------------------------------
__global__ __launch_bounds__(256)
void scan_chunk(const float* __restrict__ logF,
                const float* __restrict__ logV,
                float* __restrict__ chunkA,
                float* __restrict__ chunkL) {
    const int idx = blockIdx.x * blockDim.x + threadIdx.x;   // 0 .. B*H*NC-1
    const int h    = idx & (H - 1);
    const int rest = idx >> 10;
    const int c    = rest & (NC - 1);
    const int b    = rest >> 4;

    const size_t base = (size_t)b * T * H + (size_t)c * CL * H + h;
    const float* pF = logF + base;
    const float* pV = logV + base;

    float a = 0.f, mx = -INFINITY, sm = 0.f;
    for (int tt = 0; tt < CL; ++tt) {
        const float lf = pF[(size_t)tt * H];
        const float lv = pV[(size_t)tt * H];
        a += lf;
        const float u = lv - a;
        if (u <= mx) { sm += __expf(u - mx); }
        else         { sm = sm * __expf(mx - u) + 1.f; mx = u; }
    }
    chunkA[idx] = a;
    chunkL[idx] = mx + __logf(sm);
}

// ---------------------------------------------------------------------------
// Scan pass B: per (b,h) carry across chunks -> a0[c], R[c]; write out[b,0,h].
// ---------------------------------------------------------------------------
__global__ __launch_bounds__(256)
void scan_carry(const float* __restrict__ chunkA,
                const float* __restrict__ chunkL,
                const float* __restrict__ h0,
                float* __restrict__ a0buf,
                float* __restrict__ Rbuf,
                float* __restrict__ out) {
    const int idx = blockIdx.x * blockDim.x + threadIdx.x;   // 0 .. B*H-1
    const int h = idx & (H - 1);
    const int b = idx >> 10;

    const float lg0 = logg(h0[b * H + h]);
    out[(size_t)b * (T + 1) * H + h] = __expf(lg0);

    float a0v = 0.f, Rv = lg0;
    for (int c = 0; c < NC; ++c) {
        const size_t o = ((size_t)(b * NC + c)) * H + h;
        a0buf[o] = a0v;
        Rbuf[o]  = Rv;
        Rv = logaddexp(Rv, chunkL[o] - a0v);
        a0v += chunkA[o];
    }
}

// ---------------------------------------------------------------------------
// Scan pass C: replay chunk-local scans with carried state, emit exp(log_h).
// ---------------------------------------------------------------------------
__global__ __launch_bounds__(256)
void scan_emit(const float* __restrict__ logF,
               const float* __restrict__ logV,
               const float* __restrict__ a0buf,
               const float* __restrict__ Rbuf,
               float* __restrict__ out) {
    const int idx = blockIdx.x * blockDim.x + threadIdx.x;
    const int h    = idx & (H - 1);
    const int rest = idx >> 10;
    const int c    = rest & (NC - 1);
    const int b    = rest >> 4;

    const float a0v = a0buf[idx];
    const float Rv  = Rbuf[idx];

    const size_t base = (size_t)b * T * H + (size_t)c * CL * H + h;
    const float* pF = logF + base;
    const float* pV = logV + base;
    float* po = out + (size_t)b * (T + 1) * H + (size_t)(c * CL + 1) * H + h;

    float a = 0.f, mx = -INFINITY, sm = 0.f;
    for (int tt = 0; tt < CL; ++tt) {
        const float lf = pF[(size_t)tt * H];
        const float lv = pV[(size_t)tt * H];
        a += lf;
        const float u = lv - a;
        if (u <= mx) { sm += __expf(u - mx); }
        else         { sm = sm * __expf(mx - u) + 1.f; mx = u; }
        const float lse = logaddexp(Rv, (mx + __logf(sm)) - a0v);
        po[(size_t)tt * H] = __expf(a0v + a + lse);
    }
}

// ---------------------------------------------------------------------------
extern "C" void kernel_launch(void* const* d_in, const int* in_sizes, int n_in,
                              void* d_out, int out_size, void* d_ws, size_t ws_size,
                              hipStream_t stream) {
    const float* x   = (const float*)d_in[0];
    const float* h0  = (const float*)d_in[1];
    const float* Wf  = (const float*)d_in[2];
    const float* bf_ = (const float*)d_in[3];
    const float* Wi  = (const float*)d_in[4];
    const float* bi_ = (const float*)d_in[5];
    const float* Wh  = (const float*)d_in[6];
    const float* bh_ = (const float*)d_in[7];
    float* out = (float*)d_out;

    // workspace layout
    float* logF   = (float*)d_ws;                          // [M, H]      64 MB
    float* logV   = logF + (size_t)M * H;                  // [M, H]      64 MB
    float* chunkA = logV + (size_t)M * H;                  // [B*NC, H]
    float* chunkL = chunkA + (size_t)B * NC * H;
    float* a0buf  = chunkL + (size_t)B * NC * H;
    float* Rbuf   = a0buf  + (size_t)B * NC * H;
    unsigned short* xb  = (unsigned short*)(Rbuf + (size_t)B * NC * H);  // [M, D] bf16
    unsigned short* wfb = xb  + (size_t)M * D;                           // [H, D] bf16
    unsigned short* wib = wfb + (size_t)H * D;
    unsigned short* whb = wib + (size_t)H * D;

    // Pass 0: fp32 -> bf16 (8 elements per thread)
    convert_bf16<<<(size_t)M * D / 8 / 256, 256, 0, stream>>>(x, xb);
    convert_bf16<<<(size_t)H * D / 8 / 256, 256, 0, stream>>>(Wf, wfb);
    convert_bf16<<<(size_t)H * D / 8 / 256, 256, 0, stream>>>(Wi, wib);
    convert_bf16<<<(size_t)H * D / 8 / 256, 256, 0, stream>>>(Wh, whb);

    dim3 grid1(M / BM, H / BN);                            // 128 x 16
    minlstm_gemm_gate<<<grid1, 256, 0, stream>>>(xb, wfb, wib, whb,
                                                 bf_, bi_, bh_, logF, logV);

    scan_chunk<<<(B * H * NC) / 256, 256, 0, stream>>>(logF, logV, chunkA, chunkL);
    scan_carry<<<(B * H) / 256, 256, 0, stream>>>(chunkA, chunkL, h0, a0buf, Rbuf, out);
    scan_emit<<<(B * H * NC) / 256, 256, 0, stream>>>(logF, logV, a0buf, Rbuf, out);
}
